// AutoregressiveDeepSurvivalModel_64836826300688
// MI455X (gfx1250) — matrix operand order repaired and checked by reading, not compiled, not run on hardware
//
#include <hip/hip_runtime.h>
#include <math.h>

typedef __attribute__((ext_vector_type(16))) _Float16 v16h;
typedef __attribute__((ext_vector_type(8)))  float    v8f;

#define HH 32
#define FF 64

// fast rcp via v_rcp_f32 (1 TRANS op, ~1ulp) -- avoids the precise-div fixup chain
__device__ __forceinline__ float rcp_f(float x) { return __builtin_amdgcn_rcpf(x); }

__device__ __forceinline__ float sigmoid_f(float x) {
    return rcp_f(1.0f + __expf(-x));               // v_exp_f32 + v_rcp_f32
}
__device__ __forceinline__ float tanh_f(float x) {
    // tanh(x) = sign(x) * (1 - e) / (1 + e), e = exp(-2|x|)
    float e = __expf(-2.0f * fabsf(x));
    float r = (1.0f - e) * rcp_f(1.0f + e);
    return copysignf(r, x);
}
__device__ __forceinline__ float softplus_f(float x) {
    // stable log(1+exp(x))
    return fmaxf(x, 0.0f) + log1pf(__expf(-fabsf(x)));
}

__global__ __launch_bounds__(256)
void surv_gru_wmma_kernel(const float* __restrict__ x,
                          const float* __restrict__ t,
                          const float* __restrict__ Wenc,
                          const float* __restrict__ benc,
                          const float* __restrict__ Wih,
                          const float* __restrict__ bih,
                          const float* __restrict__ Whh,
                          const float* __restrict__ bhh,
                          const float* __restrict__ Whz,
                          const float* __restrict__ bhz,
                          const int*   __restrict__ nsp,
                          float* __restrict__ out,
                          int B)
{
    const int lane = threadIdx.x & 31;
    const int wid  = threadIdx.x >> 5;
    const int l    = lane & 15;       // column index N (batch within tile)
    const int half = lane >> 4;       // which 16-lane half
    const int tile = blockIdx.x * (blockDim.x >> 5) + wid;
    const int row0 = tile * 16 + l;   // nominal batch row
    // WMMA requires EXEC all-1s: never kill lanes; clamp and predicate the store.
    const int row  = (row0 < B) ? row0 : (B - 1);

    const int n_steps = *nsp;
    const float dtv = t[row] * rcp_f((float)n_steps);

    const int klo = half * 8;         // A-fragment K offsets per ISA layout
    const int khi = 16 + half * 8;

    // ---- A fragments: W_hh as 6 tiles of (16 gates x 32 K), f16 ----
    v16h aW[6];
    #pragma unroll
    for (int m = 0; m < 6; ++m) {
        const float* wr = Whh + (size_t)(m * 16 + l) * HH;
        #pragma unroll
        for (int q = 0; q < 4; ++q) {
            aW[m][2*q+0]   = (_Float16)wr[klo + 2*q + 0];
            aW[m][2*q+1]   = (_Float16)wr[klo + 2*q + 1];
            aW[m][8+2*q+0] = (_Float16)wr[khi + 2*q + 0];
            aW[m][8+2*q+1] = (_Float16)wr[khi + 2*q + 1];
        }
    }

    // ---- per-lane gate constants ----
    // r/z tiles (m=0..3): fold gx + b_hh into the WMMA C input
    v8f crz[4];
    #pragma unroll
    for (int m = 0; m < 4; ++m)
        #pragma unroll
        for (int r = 0; r < 8; ++r) {
            int g = 16 * m + half * 8 + r;
            crz[m][r] = dtv * Wih[g] + bih[g] + bhh[g];
        }
    // n tiles (m=4,5): only b_hh goes inside gh_n (it is scaled by r);
    // gx_n is added after r*gh_n, keep it separate.
    v8f   cnb[2];
    float gxn[16];
    #pragma unroll
    for (int m = 0; m < 2; ++m)
        #pragma unroll
        for (int r = 0; r < 8; ++r) {
            int g = 64 + 16 * m + half * 8 + r;
            cnb[m][r]    = bhh[g];
            gxn[8*m + r] = dtv * Wih[g] + bih[g];
        }
    // hazard weights, aligned to local feature map f(j) = (j<8 ? half*8+j : 16+half*8+j-8)
    float whz[16];
    #pragma unroll
    for (int j = 0; j < 8; ++j) {
        whz[j]   = Whz[half * 8 + j];
        whz[8+j] = Whz[16 + half * 8 + j];
    }
    const float bhz0 = bhz[0];

    // ---- encoder: h0^T = W_enc @ x^T  (2 feature tiles x 2 K-halves) ----
    v8f c0, c1;
    #pragma unroll
    for (int r = 0; r < 8; ++r) {
        c0[r] = benc[half * 8 + r];
        c1[r] = benc[16 + half * 8 + r];
    }
    #pragma unroll
    for (int kh = 0; kh < 2; ++kh) {
        // B fragment: column N=l, K = kh*32 + (i<8 ? half*8+i : 16+half*8+i-8)
        const float* xr = x + (size_t)row * FF + kh * 32 + half * 8;
        v16h bx;
        #pragma unroll
        for (int i = 0; i < 16; ++i) bx[i] = (_Float16)xr[(i < 8) ? i : i + 8];

        const float* w0 = Wenc + (size_t)(0 * 16 + l) * FF + kh * 32;
        const float* w1 = Wenc + (size_t)(1 * 16 + l) * FF + kh * 32;
        v16h a0, a1;
        #pragma unroll
        for (int q = 0; q < 4; ++q) {
            a0[2*q+0]   = (_Float16)w0[klo + 2*q + 0];
            a0[2*q+1]   = (_Float16)w0[klo + 2*q + 1];
            a0[8+2*q+0] = (_Float16)w0[khi + 2*q + 0];
            a0[8+2*q+1] = (_Float16)w0[khi + 2*q + 1];
            a1[2*q+0]   = (_Float16)w1[klo + 2*q + 0];
            a1[2*q+1]   = (_Float16)w1[klo + 2*q + 1];
            a1[8+2*q+0] = (_Float16)w1[khi + 2*q + 0];
            a1[8+2*q+1] = (_Float16)w1[khi + 2*q + 1];
        }
        c0 = __builtin_amdgcn_wmma_f32_16x16x32_f16(false, a0, false, bx, (short)0, c0, false, false);
        c1 = __builtin_amdgcn_wmma_f32_16x16x32_f16(false, a1, false, bx, (short)0, c1, false, false);
    }

    // local h layout: h[j] = feature f(j) of batch `row`
    float h[16];
    #pragma unroll
    for (int r = 0; r < 8; ++r) { h[r] = c0[r]; h[8+r] = c1[r]; }

    // ---- 100 recurrent steps, everything register-resident ----
    float integ = 0.0f, hz = 0.0f;
    for (int s = 0; s < n_steps; ++s) {
        // Build B fragment (h^T, 32x16): lane = column N=l, K = f(i).
        // Local h layout already matches; no swap between lane pairs needed.
        v16h bh;
        #pragma unroll
        for (int j = 0; j < 8; ++j) {
            float o0 = __shfl_xor(h[j],     16, 32);  // other half's features
            float o1 = __shfl_xor(h[j + 8], 16, 32);
            float lo = h[j];                          // element i=j
            float hi = h[j + 8];                      // element i=8+j
            bh[j]     = (_Float16)lo;
            bh[8 + j] = (_Float16)hi;
        }

        // gh (+ folded biases): 6 x v_wmma_f32_16x16x32_f16, K=32 in one shot
        v8f c[6];
        c[0] = __builtin_amdgcn_wmma_f32_16x16x32_f16(false, aW[0], false, bh, (short)0, crz[0], false, false);
        c[1] = __builtin_amdgcn_wmma_f32_16x16x32_f16(false, aW[1], false, bh, (short)0, crz[1], false, false);
        c[2] = __builtin_amdgcn_wmma_f32_16x16x32_f16(false, aW[2], false, bh, (short)0, crz[2], false, false);
        c[3] = __builtin_amdgcn_wmma_f32_16x16x32_f16(false, aW[3], false, bh, (short)0, crz[3], false, false);
        c[4] = __builtin_amdgcn_wmma_f32_16x16x32_f16(false, aW[4], false, bh, (short)0, cnb[0], false, false);
        c[5] = __builtin_amdgcn_wmma_f32_16x16x32_f16(false, aW[5], false, bh, (short)0, cnb[1], false, false);

        // gates: r/z/n for feature f(j) all live in this lane (exp+rcp forms,
        // TRANS ops co-execute with the XDL WMMAs)
        float dot = 0.0f;
        #pragma unroll
        for (int j = 0; j < 16; ++j) {
            int p = j >> 3, r = j & 7;
            float rr = sigmoid_f(c[p][r]);
            float zz = sigmoid_f(c[2 + p][r]);
            float nn = tanh_f(gxn[j] + rr * c[4 + p][r]);
            h[j] = (1.0f - zz) * nn + zz * h[j];
            dot += h[j] * whz[j];
        }
        // hazard: combine the two half-dots across the lane pair
        dot += __shfl_xor(dot, 16, 32);
        hz = softplus_f(dot + bhz0);
        integ += hz * dtv;
    }

    if (half == 0 && row0 < B) {
        out[row]     = integ;  // integrated hazard
        out[B + row] = hz;     // hazard at final step
    }
}

extern "C" void kernel_launch(void* const* d_in, const int* in_sizes, int n_in,
                              void* d_out, int out_size, void* d_ws, size_t ws_size,
                              hipStream_t stream) {
    const float* x    = (const float*)d_in[0];
    const float* t    = (const float*)d_in[1];
    const float* Wenc = (const float*)d_in[2];
    const float* benc = (const float*)d_in[3];
    const float* Wih  = (const float*)d_in[4];
    const float* bih  = (const float*)d_in[5];
    const float* Whh  = (const float*)d_in[6];
    const float* bhh  = (const float*)d_in[7];
    const float* Whz  = (const float*)d_in[8];
    const float* bhz  = (const float*)d_in[9];
    const int*   nsp  = (const int*)d_in[10];
    float* out = (float*)d_out;

    const int B = in_sizes[1];            // t has B elements
    const int threads = 256;              // 8 waves -> 128 batch rows per block
    const int rowsPerBlock = (threads / 32) * 16;
    const int blocks = (B + rowsPerBlock - 1) / rowsPerBlock;

    surv_gru_wmma_kernel<<<blocks, threads, 0, stream>>>(
        x, t, Wenc, benc, Wih, bih, Whh, bhh, Whz, bhz, nsp, out, B);
}
